// MoELayer_37391985279403
// MI455X (gfx1250) — compile-verified, matrix operand order checked by reading, not run on hardware
//
#include <hip/hip_runtime.h>

// MoE SwiGLU layer for MI455X (gfx1250, wave32, WMMA + TDM).
// B=4, T=2048 -> N=8192 tokens, D=1024, D_FF=4096, E=8, top-2.
//
// Workspace layout (assumes ws_size >= ~85 MB):
//   xb   : bf16 [N, D]      (16.8 MB)
//   hb   : bf16 [N, D_FF]   (67.1 MB)   reused per expert
//   comb : f32  [N, E]      (0.26 MB)

#define DIM_D   1024
#define DIM_FF  4096
#define NEXP    8

typedef __attribute__((ext_vector_type(16))) __bf16       v16bf;
typedef __attribute__((ext_vector_type(8)))  float        v8f;
typedef __attribute__((ext_vector_type(4))) unsigned int  u32x4;
typedef __attribute__((ext_vector_type(2))) unsigned int  u32x2;
typedef __attribute__((ext_vector_type(4)))  float        f32x4;
typedef __attribute__((ext_vector_type(8)))  int          i32x8;
typedef __attribute__((ext_vector_type(4)))  int          i32x4;

#if defined(__HIP_DEVICE_COMPILE__)
# if __has_builtin(__builtin_amdgcn_tensor_load_to_lds) && \
     __has_builtin(__builtin_amdgcn_s_wait_tensorcnt)
#  define USE_TDM 1
# endif
# if __has_builtin(__builtin_amdgcn_cvt_pk_bf16_f32)
#  define USE_CVT_PK 1
# endif
#endif

union FragU { v16bf v; u32x4 q[2]; };

// Scalar fp32 -> bf16, round-half-up (2 VALU: add + shift).
__device__ __forceinline__ unsigned short f2bf(float f) {
    return (unsigned short)((__float_as_uint(f) + 0x8000u) >> 16);
}

// Packed pair fp32 -> bf16x2.
// Preferred: v_cvt_pk_bf16_f32 (1 op). Fallback: round-half-up adds + one
// v_perm_b32 picking both high halves (3 VALU per pair).
__device__ __forceinline__ unsigned pack_bf16x2(float a, float b) {
#if defined(USE_CVT_PK)
    typedef __attribute__((ext_vector_type(2))) __bf16 v2bf;
    v2bf r = __builtin_amdgcn_cvt_pk_bf16_f32(a, b);
    return __builtin_bit_cast(unsigned, r);
#else
    unsigned ra = __float_as_uint(a) + 0x8000u;
    unsigned rb = __float_as_uint(b) + 0x8000u;
    // result bytes: [1:0] = ra[3:2], [3:2] = rb[3:2]
    return __builtin_amdgcn_perm(rb, ra, 0x07060302u);
#endif
}

// Load one 16x32 bf16 operand fragment from a K-major LDS tile.
// ISA layout: lanes 0-15 hold row (lane), K 0..7 / 16..23;
//             lanes 16-31 hold row (lane&15), K 8..15 / 24..31.
__device__ __forceinline__ v16bf load_frag(const unsigned short* row_base) {
    FragU f;
    f.q[0] = *(const u32x4*)(row_base);        // K 0-7   (or 8-15)
    f.q[1] = *(const u32x4*)(row_base + 16);   // K 16-23 (or 24-31)
    return f.v;
}

#define LDSS 40   // padded LDS row stride (elems); 80B keeps 16B alignment

#if defined(USE_TDM)
// TDM copy of one bf16 A-tile (128 rows x 32 elems) global -> LDS.
// LDS padding done by the TDM: pad_interval=3 (every 16 DWORDs = 64B row),
// pad_amount=3 (4 DWORDs = 16B) -> row stride 80B = LDSS elems.
__device__ __forceinline__ void tdm_load_a(unsigned lds_byte_addr,
                                           const void* gaddr,
                                           unsigned tensor_dim0,        // elems/row
                                           unsigned long long row_stride_elems) {
    unsigned long long ga = (unsigned long long)(uintptr_t)gaddr;
    u32x4 g0;
    g0.x = 1u;                                             // count=1 (valid), user mode
    g0.y = lds_byte_addr;                                  // lds_addr [63:32]
    g0.z = (unsigned)(ga & 0xffffffffu);                   // global_addr [95:64]
    g0.w = (unsigned)((ga >> 32) & 0x1ffffffu) | (2u << 30); // addr msbs + type=2
    i32x8 g1;
    g1[0] = (1 << 16) | (1 << 20) | (3 << 22) | (3 << 25); // data=2B, pad en/intvl/amt
    g1[1] = (int)((tensor_dim0 & 0xffffu) << 16);          // tensor_dim0 lo16
    g1[2] = (int)((tensor_dim0 >> 16) & 0xffffu);          // tensor_dim0 hi16, dim1 lo=0
    g1[3] = (int)(0x10u | (32u << 16));                    // tensor_dim1 hi16, tile_dim0=32
    g1[4] = (int)128;                                      // tile_dim1=128, tile_dim2=0
    g1[5] = (int)(row_stride_elems & 0xffffffffu);         // tensor_dim0_stride lo32
    g1[6] = (int)((row_stride_elems >> 32) & 0xffffu);     // stride hi16, dim1_stride=0
    g1[7] = 0;
    i32x4 z4 = {0, 0, 0, 0};
    i32x8 z8 = {0, 0, 0, 0, 0, 0, 0, 0};
    __builtin_amdgcn_tensor_load_to_lds(g0, g1, z4, z4, z8, 0);
}
#endif

// ---------------------------------------------------------------- convert x
__global__ __launch_bounds__(256)
void moe_convert_x(const float* __restrict__ x, unsigned short* __restrict__ xb,
                   size_t n) {
    size_t i = ((size_t)blockIdx.x * 256 + threadIdx.x) * 4;
    if (i >= n) return;
    f32x4 v = *(const f32x4*)&x[i];
    u32x2 p;
    p.x = pack_bf16x2(v.x, v.y);
    p.y = pack_bf16x2(v.z, v.w);
    *(u32x2*)&xb[i] = p;
}

// ------------------------------------------------------------------- router
__global__ __launch_bounds__(256)
void moe_router(const float* __restrict__ x, const float* __restrict__ gw,
                float* __restrict__ comb, int n_tok) {
    int tok  = blockIdx.x * 8 + (threadIdx.x >> 5);
    int lane = threadIdx.x & 31;
    if (tok >= n_tok) return;

    float acc[NEXP];
#pragma unroll
    for (int e = 0; e < NEXP; ++e) acc[e] = 0.f;

    const float* xp = x + (size_t)tok * DIM_D;
    for (int c = lane; c < DIM_D; c += 32) {
        float xv = xp[c];
#pragma unroll
        for (int e = 0; e < NEXP; ++e) acc[e] += xv * gw[e * DIM_D + c];
    }
#pragma unroll
    for (int e = 0; e < NEXP; ++e)
#pragma unroll
        for (int off = 16; off > 0; off >>= 1)
            acc[e] += __shfl_xor(acc[e], off, 32);

    if (lane == 0) {
        int i1 = 0; float v1 = acc[0];
#pragma unroll
        for (int e = 1; e < NEXP; ++e) if (acc[e] > v1) { v1 = acc[e]; i1 = e; }
        int i2 = -1; float v2 = -3.0e38f;
#pragma unroll
        for (int e = 0; e < NEXP; ++e)
            if (e != i1 && acc[e] > v2) { v2 = acc[e]; i2 = e; }
        float e2 = __expf(v2 - v1);          // v1 is the max
        float inv = 1.f / (1.f + e2);
#pragma unroll
        for (int e = 0; e < NEXP; ++e)
            comb[(size_t)tok * NEXP + e] =
                (e == i1) ? inv : ((e == i2) ? e2 * inv : 0.f);
    }
}

// -------------------------------------------------- GEMM1: h = silu(xWg)*xW1
// Macro tile 128 tokens x 64 ffn cols, K-step 32, 8 waves = 4(M) x 2(N).
__global__ __launch_bounds__(256)
void moe_gemm1(const unsigned short* __restrict__ xb,   // [N, D] bf16
               const float* __restrict__ w1,            // [E, FF, D]
               const float* __restrict__ wg,            // [E, FF, D]
               unsigned short* __restrict__ hb,         // [N, FF] bf16
               int expert) {
    __shared__ unsigned short sA[128 * LDSS];
    __shared__ unsigned short sG[64 * LDSS];
    __shared__ unsigned short sW[64 * LDSS];

    const int tid  = threadIdx.x;
    const int lane = tid & 31;
    const int wave = tid >> 5;
    const int wm   = wave >> 1;          // 0..3 : 32-token block
    const int wn   = wave & 1;           // 0..1 : 32-col block
    const int tok0 = blockIdx.y * 128;
    const int col0 = blockIdx.x * 64;

    const float* wgp = wg + (size_t)expert * DIM_FF * DIM_D;
    const float* w1p = w1 + (size_t)expert * DIM_FF * DIM_D;

    const v8f vzero = {0.f, 0.f, 0.f, 0.f, 0.f, 0.f, 0.f, 0.f};
    v8f accG[2][2], accW[2][2];
#pragma unroll
    for (int mt = 0; mt < 2; ++mt)
#pragma unroll
        for (int nt = 0; nt < 2; ++nt) { accG[mt][nt] = vzero; accW[mt][nt] = vzero; }

    for (int k0 = 0; k0 < DIM_D; k0 += 32) {
        // A tile: 128x32 bf16 straight copy via Tensor Data Mover
#if defined(USE_TDM)
        if (wave == 0)
            tdm_load_a((unsigned)(uintptr_t)(void*)sA,
                       &xb[(size_t)tok0 * DIM_D + k0], DIM_D, DIM_D);
#else
#pragma unroll
        for (int i = 0; i < 2; ++i) {
            int idx = tid + i * 256;
            int r = idx >> 2, c = (idx & 3) * 8;
            *(u32x4*)&sA[r * LDSS + c] =
                *(const u32x4*)&xb[(size_t)(tok0 + r) * DIM_D + k0 + c];
        }
#endif
        // W tiles: 64x32 fp32 -> bf16 (both matrices)
#pragma unroll
        for (int i = 0; i < 2; ++i) {
            int idx = tid + i * 256;
            int r = idx >> 3, c = (idx & 7) * 4;
            size_t go = (size_t)(col0 + r) * DIM_D + k0 + c;
            f32x4 g = *(const f32x4*)&wgp[go];
            f32x4 w = *(const f32x4*)&w1p[go];
            u32x2 pg, pw;
            pg.x = pack_bf16x2(g.x, g.y);
            pg.y = pack_bf16x2(g.z, g.w);
            pw.x = pack_bf16x2(w.x, w.y);
            pw.y = pack_bf16x2(w.z, w.w);
            *(u32x2*)&sG[r * LDSS + c] = pg;
            *(u32x2*)&sW[r * LDSS + c] = pw;
        }
        if (k0 + 32 < DIM_D) {
            __builtin_prefetch(&wgp[(size_t)col0 * DIM_D + k0 + 32], 0, 1);
            __builtin_prefetch(&w1p[(size_t)col0 * DIM_D + k0 + 32], 0, 1);
        }
#if defined(USE_TDM)
        if (wave == 0)
            __builtin_amdgcn_s_wait_tensorcnt(0);
#endif
        __syncthreads();

        const int rsel  = (lane & 15);
        const int khalf = (lane >> 4) * 8;
        v16bf a[2], bg[2], bw[2];
#pragma unroll
        for (int mt = 0; mt < 2; ++mt)
            a[mt] = load_frag(&sA[(wm * 32 + mt * 16 + rsel) * LDSS + khalf]);
#pragma unroll
        for (int nt = 0; nt < 2; ++nt) {
            bg[nt] = load_frag(&sG[(wn * 32 + nt * 16 + rsel) * LDSS + khalf]);
            bw[nt] = load_frag(&sW[(wn * 32 + nt * 16 + rsel) * LDSS + khalf]);
        }
#pragma unroll
        for (int mt = 0; mt < 2; ++mt)
#pragma unroll
            for (int nt = 0; nt < 2; ++nt) {
                accG[mt][nt] = __builtin_amdgcn_wmma_f32_16x16x32_bf16(
                    false, a[mt], false, bg[nt], (short)0, accG[mt][nt], false, false);
                accW[mt][nt] = __builtin_amdgcn_wmma_f32_16x16x32_bf16(
                    false, a[mt], false, bw[nt], (short)0, accW[mt][nt], false, false);
            }
        __syncthreads();
    }

    // Epilogue: h = silu(g) * w, store bf16.
    // C layout: lane holds col = lane&15; VGPR r holds M = r + 8*(lane>>4).
#pragma unroll
    for (int mt = 0; mt < 2; ++mt)
#pragma unroll
        for (int nt = 0; nt < 2; ++nt) {
            int col = col0 + wn * 32 + nt * 16 + (lane & 15);
#pragma unroll
            for (int r = 0; r < 8; ++r) {
                int tok = tok0 + wm * 32 + mt * 16 + (lane >> 4) * 8 + r;
                float g = accG[mt][nt][r];
                float w = accW[mt][nt][r];
                float h = (g / (1.f + __expf(-g))) * w;
                hb[(size_t)tok * DIM_FF + col] = f2bf(h);
            }
        }
}

// ----------------------------------------------- GEMM2: out += comb * h W2^T
__global__ __launch_bounds__(256)
void moe_gemm2(const unsigned short* __restrict__ hb,   // [N, FF] bf16
               const float* __restrict__ w2,            // [E, D, FF]
               const float* __restrict__ comb,          // [N, E]
               float* __restrict__ out,                 // [N, D]
               int expert, int accumulate) {
    __shared__ unsigned short sA[128 * LDSS];
    __shared__ unsigned short sB[64 * LDSS];
    __shared__ float sC[128];

    const int tid  = threadIdx.x;
    const int lane = tid & 31;
    const int wave = tid >> 5;
    const int wm   = wave >> 1;
    const int wn   = wave & 1;
    const int tok0 = blockIdx.y * 128;
    const int col0 = blockIdx.x * 64;

    const float* w2p = w2 + (size_t)expert * DIM_D * DIM_FF;

    if (tid < 128)
        sC[tid] = comb[(size_t)(tok0 + tid) * NEXP + expert];

    const v8f vzero = {0.f, 0.f, 0.f, 0.f, 0.f, 0.f, 0.f, 0.f};
    v8f acc[2][2];
#pragma unroll
    for (int mt = 0; mt < 2; ++mt)
#pragma unroll
        for (int nt = 0; nt < 2; ++nt) acc[mt][nt] = vzero;

    for (int k0 = 0; k0 < DIM_FF; k0 += 32) {
#if defined(USE_TDM)
        if (wave == 0)
            tdm_load_a((unsigned)(uintptr_t)(void*)sA,
                       &hb[(size_t)tok0 * DIM_FF + k0], DIM_FF, DIM_FF);
#else
#pragma unroll
        for (int i = 0; i < 2; ++i) {
            int idx = tid + i * 256;
            int r = idx >> 2, c = (idx & 3) * 8;
            *(u32x4*)&sA[r * LDSS + c] =
                *(const u32x4*)&hb[(size_t)(tok0 + r) * DIM_FF + k0 + c];
        }
#endif
#pragma unroll
        for (int i = 0; i < 2; ++i) {
            int idx = tid + i * 256;
            int r = idx >> 3, c = (idx & 7) * 4;
            f32x4 w = *(const f32x4*)&w2p[(size_t)(col0 + r) * DIM_FF + k0 + c];
            u32x2 pw;
            pw.x = pack_bf16x2(w.x, w.y);
            pw.y = pack_bf16x2(w.z, w.w);
            *(u32x2*)&sB[r * LDSS + c] = pw;
        }
        if (k0 + 32 < DIM_FF)
            __builtin_prefetch(&w2p[(size_t)col0 * DIM_FF + k0 + 32], 0, 1);
#if defined(USE_TDM)
        if (wave == 0)
            __builtin_amdgcn_s_wait_tensorcnt(0);
#endif
        __syncthreads();

        const int rsel  = (lane & 15);
        const int khalf = (lane >> 4) * 8;
        v16bf a[2], b[2];
#pragma unroll
        for (int mt = 0; mt < 2; ++mt)
            a[mt] = load_frag(&sA[(wm * 32 + mt * 16 + rsel) * LDSS + khalf]);
#pragma unroll
        for (int nt = 0; nt < 2; ++nt)
            b[nt] = load_frag(&sB[(wn * 32 + nt * 16 + rsel) * LDSS + khalf]);
#pragma unroll
        for (int mt = 0; mt < 2; ++mt)
#pragma unroll
            for (int nt = 0; nt < 2; ++nt)
                acc[mt][nt] = __builtin_amdgcn_wmma_f32_16x16x32_bf16(
                    false, a[mt], false, b[nt], (short)0, acc[mt][nt], false, false);
        __syncthreads();
    }

#pragma unroll
    for (int mt = 0; mt < 2; ++mt)
#pragma unroll
        for (int nt = 0; nt < 2; ++nt) {
            int col = col0 + wn * 32 + nt * 16 + (lane & 15);
#pragma unroll
            for (int r = 0; r < 8; ++r) {
                int mloc = wm * 32 + mt * 16 + (lane >> 4) * 8 + r;
                int tok  = tok0 + mloc;
                float v  = acc[mt][nt][r] * sC[mloc];
                size_t o = (size_t)tok * DIM_D + col;
                out[o] = accumulate ? (out[o] + v) : v;
            }
        }
}

// ---------------------------------------------------------------- launcher
extern "C" void kernel_launch(void* const* d_in, const int* in_sizes, int n_in,
                              void* d_out, int out_size, void* d_ws, size_t ws_size,
                              hipStream_t stream) {
    const float* x  = (const float*)d_in[0];   // [B,T,D]
    const float* gw = (const float*)d_in[1];   // [E,D]
    const float* w1 = (const float*)d_in[2];   // [E,FF,D]
    const float* wg = (const float*)d_in[3];   // [E,FF,D]
    const float* w2 = (const float*)d_in[4];   // [E,D,FF]
    float* out = (float*)d_out;

    const int n_tok = in_sizes[0] / DIM_D;     // 8192

    char* ws = (char*)d_ws;
    unsigned short* xb = (unsigned short*)ws;
    unsigned short* hb = (unsigned short*)(ws + (size_t)n_tok * DIM_D * 2);
    float* comb = (float*)(ws + (size_t)n_tok * DIM_D * 2 +
                                (size_t)n_tok * DIM_FF * 2);

    size_t nx = (size_t)n_tok * DIM_D;
    moe_convert_x<<<dim3((unsigned)((nx / 4 + 255) / 256)), 256, 0, stream>>>(x, xb, nx);
    moe_router<<<dim3((n_tok + 7) / 8), 256, 0, stream>>>(x, gw, comb, n_tok);

    for (int e = 0; e < NEXP; ++e) {
        moe_gemm1<<<dim3(DIM_FF / 64, n_tok / 128), 256, 0, stream>>>(xb, w1, wg, hb, e);
        moe_gemm2<<<dim3(DIM_D / 64, n_tok / 128), 256, 0, stream>>>(hb, w2, comb, out,
                                                                     e, e > 0 ? 1 : 0);
    }
}
